// StrongerEdgeGNN_81793357185762
// MI455X (gfx1250) — matrix operand order repaired and checked by reading, not compiled, not run on hardware
//
#include <hip/hip_runtime.h>
#include <hip/hip_bf16.h>

// ---------------------------------------------------------------------------
// StrongerEdgeGNN forward on MI455X (gfx1250, wave32, WMMA + TDM).
// Dense linears: f16-input / f32-accumulate WMMA (v_wmma_f32_16x16x32_f16).
// Weight-stationary: weights pre-swizzled to f16 fragment order, B fragments
// read as coalesced global_load_b128 from WGP$/L2.  For K=128 GEMMs the whole
// 32-row A tile is loaded by ONE Tensor-Data-Mover descriptor
// (TENSOR_LOAD_TO_LDS, padded rows for bank-conflict-free fragment reads),
// waited with s_wait_tensorcnt; the inner loop then runs barrier-free.
// d_out: [0] = loss, [1..256] = logits (64 x 4, row-major), float32.
// ---------------------------------------------------------------------------

typedef __attribute__((ext_vector_type(16))) _Float16 v16h;
typedef __attribute__((ext_vector_type(8)))  _Float16 v8h;
typedef __attribute__((ext_vector_type(4)))  _Float16 v4h;
typedef __attribute__((ext_vector_type(8)))  float    v8f;

#if defined(__has_builtin)
#if __has_builtin(__builtin_amdgcn_tensor_load_to_lds) && \
    __has_builtin(__builtin_amdgcn_s_wait_tensorcnt)
#define SEGNN_HAS_TDM 1
#endif
#endif
#ifndef SEGNN_HAS_TDM
#define SEGNN_HAS_TDM 0
#endif

#if SEGNN_HAS_TDM
typedef __attribute__((ext_vector_type(4))) unsigned int u32x4;
typedef __attribute__((ext_vector_type(8))) int          i32x8;
typedef __attribute__((ext_vector_type(4))) int          i32x4;
#endif

#define G_CONST   64
#define NG_CONST  512
#define NN_CONST  32768          // nodes
#define E_CONST   131072         // edges
#define KPAD      832            // 831 padded to /32

__device__ inline float wave_reduce_sum(float v) {
#pragma unroll
  for (int off = 16; off > 0; off >>= 1) v += __shfl_xor(v, off, 32);
  return v;
}

__device__ inline void atomicMaxFloat(float* addr, float val) {
  int* ai = (int*)addr;
  int old = __float_as_int(*addr);
  while (__int_as_float(old) < val) {
    int assumed = old;
    old = atomicCAS(ai, assumed, __float_as_int(val));
    if (old == assumed) break;
  }
}

__device__ inline float logsigf(float x) {
  return (x >= 0.0f) ? -log1pf(__expf(-x)) : (x - log1pf(__expf(x)));
}

// ---------------------------------------------------------------------------
// Weight pre-swizzle: W (f32, row-major KxN, logical K rows = Ksrc, padded to
// Kdst%32==0) -> f16 fragment order [kt][nt][lane 0..31][h 0..15] where
//   k = kt*32 + (lane>>4)*8 + (h&7) + (h>>3)*16,  n = nt*16 + (lane&15)
// ---------------------------------------------------------------------------
__global__ void segnn_swizzle_w(const float* __restrict__ W, _Float16* __restrict__ o,
                                int Ksrc, int N, size_t tot)
{
  size_t i = (size_t)blockIdx.x * blockDim.x + threadIdx.x;
  if (i >= tot) return;
  int h    = (int)(i & 15);
  int lane = (int)((i >> 4) & 31);
  size_t tile = i >> 9;
  int ntiles = N >> 4;
  int kt = (int)(tile / ntiles), nt = (int)(tile % ntiles);
  int k = kt * 32 + ((lane >> 4) << 3) + (h & 7) + ((h >> 3) << 4);
  int n = (nt << 4) + (lane & 15);
  o[i] = (k < Ksrc) ? (_Float16)W[(size_t)k * N + n] : (_Float16)0.0f;
}

// ---------------------------------------------------------------------------
// Generic GEMM (K % 32 == 0, any K): LDS-staged f16 A tile.
// ---------------------------------------------------------------------------
__global__ __launch_bounds__(256) void segnn_gemm_wmma(
    const float* __restrict__ A, const _Float16* __restrict__ Wsw,
    const float* __restrict__ bias, float* __restrict__ C,
    int M, int K, int N, int act)
{
  __shared__ _Float16 As[32][40];
  const int tid  = threadIdx.x;
  const int wave = tid >> 5;
  const int lane = tid & 31;

  const int tilesN  = N >> 7;
  const int bm      = blockIdx.x / tilesN;
  const int bn      = blockIdx.x % tilesN;
  const int rowBase = bm * 32;
  const int colBase = bn * 128;

  const int wm = (wave >> 2) * 16;
  const int wn = (wave & 3) * 32;

  const int ar = tid >> 3, ac = (tid & 7) * 4;
  const int mA     = lane & 15;
  const int kHalfA = (lane >> 4) * 8;
  const int ntiles = N >> 4;
  const int nt0    = (colBase >> 4) + (wn >> 4);

  v8f acc0 = {}, acc1 = {};

  for (int k0 = 0; k0 < K; k0 += 32) {
    {
      const float4 va = *(const float4*)(A + (size_t)(rowBase + ar) * K + k0 + ac);
      v4h h;
      h[0] = (_Float16)va.x; h[1] = (_Float16)va.y;
      h[2] = (_Float16)va.z; h[3] = (_Float16)va.w;
      *(v4h*)&As[ar][ac] = h;
    }
    if (k0 + 32 < K)
      __builtin_prefetch((const void*)(A + (size_t)(rowBase + ar) * K + (k0 + 32) + ac), 0, 1);
    __syncthreads();

    const int kt = k0 >> 5;
    const _Float16* bp0 = Wsw + (((size_t)kt * ntiles + nt0) << 9) + lane * 16;
    const v8h b0lo = *(const v8h*)bp0;
    const v8h b0hi = *(const v8h*)(bp0 + 8);
    const v8h b1lo = *(const v8h*)(bp0 + 512);
    const v8h b1hi = *(const v8h*)(bp0 + 520);

    const v8h alo = *(const v8h*)&As[wm + mA][kHalfA];
    const v8h ahi = *(const v8h*)&As[wm + mA][kHalfA + 16];

    v16h a  = __builtin_shufflevector(alo, ahi, 0,1,2,3,4,5,6,7,8,9,10,11,12,13,14,15);
    v16h b0 = __builtin_shufflevector(b0lo, b0hi, 0,1,2,3,4,5,6,7,8,9,10,11,12,13,14,15);
    v16h b1 = __builtin_shufflevector(b1lo, b1hi, 0,1,2,3,4,5,6,7,8,9,10,11,12,13,14,15);

    acc0 = __builtin_amdgcn_wmma_f32_16x16x32_f16(
        false, a, false, b0, (short)0, acc0, false, false);
    acc1 = __builtin_amdgcn_wmma_f32_16x16x32_f16(
        false, a, false, b1, (short)0, acc1, false, false);
    __syncthreads();
  }

  const int n0 = colBase + wn + (lane & 15);
  const int mb = (lane >> 4) * 8;
  const float bv0 = bias ? bias[n0] : 0.0f;
  const float bv1 = bias ? bias[n0 + 16] : 0.0f;
#pragma unroll
  for (int r = 0; r < 8; ++r) {
    int m = rowBase + wm + mb + r;
    float v0 = acc0[r] + bv0;
    float v1 = acc1[r] + bv1;
    if (act == 1)      { v0 = fmaxf(v0, 0.0f); v1 = fmaxf(v1, 0.0f); }
    else if (act == 2) { v0 = 1.0f / (1.0f + __expf(-v0));
                         v1 = 1.0f / (1.0f + __expf(-v1)); }
    C[(size_t)m * N + n0]      = v0;
    C[(size_t)m * N + n0 + 16] = v1;
  }
}

// ---------------------------------------------------------------------------
// K=128 specialized GEMM: one TDM descriptor loads the full 32x128 f32 A tile
// into LDS with 4-dword row padding (stride 132 dwords -> conflict-free reads).
// Inner loop (4 unrolled k-steps) is barrier-free.
// ---------------------------------------------------------------------------
__global__ __launch_bounds__(256) void segnn_gemm_wmma_k128(
    const float* __restrict__ A, const _Float16* __restrict__ Wsw,
    const float* __restrict__ bias, float* __restrict__ C,
    int M, int N, int act)
{
  __shared__ float AsF[32 * 132];   // 32 rows x (128 + 4 pad) f32
  const int tid  = threadIdx.x;
  const int wave = tid >> 5;
  const int lane = tid & 31;

  const int tilesN  = N >> 7;
  const int bm      = blockIdx.x / tilesN;
  const int bn      = blockIdx.x % tilesN;
  const int rowBase = bm * 32;
  const int colBase = bn * 128;

  const int wm = (wave >> 2) * 16;
  const int wn = (wave & 3) * 32;
  const int mA     = lane & 15;
  const int kHalfA = (lane >> 4) * 8;
  const int ntiles = N >> 4;
  const int nt0    = (colBase >> 4) + (wn >> 4);

#if SEGNN_HAS_TDM
  if (wave == 0) {
    unsigned long long ga = (unsigned long long)(uintptr_t)(A + (size_t)rowBase * 128);
    unsigned int lds = (unsigned int)(uintptr_t)&AsF[0];
    u32x4 g0;
    g0[0] = 1u;                                        // count=1 (valid D#)
    g0[1] = lds;                                       // LDS byte address
    g0[2] = (unsigned int)ga;                          // global_addr[31:0]
    g0[3] = ((unsigned int)(ga >> 32) & 0x01FFFFFFu)   // global_addr[56:32]
            | (2u << 30);                              // type=2 ("image")
    i32x8 g1;
    g1[0] = (int)((2u << 16)      // data_size = 4 bytes
                | (1u << 20)      // pad_enable
                | (6u << 22)      // pad_interval: 128 dwords
                | (3u << 25));    // pad_amount: 4 dwords
    g1[1] = (int)(128u << 16);                 // tensor_dim0[15:0] = 128
    g1[2] = (int)(32u << 16);                  // tensor_dim0 hi=0 | tensor_dim1[15:0]=32
    g1[3] = (int)(128u << 16);                 // tensor_dim1 hi=0 | tile_dim0 = 128
    g1[4] = (int)32u;                          // tile_dim1 = 32, tile_dim2 = 0
    g1[5] = (int)128u;                         // tensor_dim0_stride lo = 128 elems
    g1[6] = 0;                                 // stride0 hi | stride1 lo
    g1[7] = 0;                                 // stride1 hi
    i32x4 z4 = {0, 0, 0, 0};
#if __has_include(<hip/amd_detail/amd_gfx1250_TDM.h>)
    i32x8 z8 = {0, 0, 0, 0, 0, 0, 0, 0};
    __builtin_amdgcn_tensor_load_to_lds(g0, g1, z4, z4, z8, 0);
#else
    __builtin_amdgcn_tensor_load_to_lds(g0, g1, z4, z4, 0);
#endif
    __builtin_amdgcn_s_wait_tensorcnt(0);
  }
  __syncthreads();
#else
  // fallback: cooperative vector-load staging into the same padded layout
  for (int i = tid; i < 32 * 32; i += 256) {
    int r = i >> 5, c4 = (i & 31) * 4;
    *(float4*)&AsF[r * 132 + c4] =
        *(const float4*)(A + (size_t)(rowBase + r) * 128 + c4);
  }
  __syncthreads();
#endif

  v8f acc0 = {}, acc1 = {};
  const float* arow = &AsF[(wm + mA) * 132];

#pragma unroll
  for (int kt = 0; kt < 4; ++kt) {
    const int k0 = kt * 32;
    const _Float16* bp0 = Wsw + (((size_t)kt * ntiles + nt0) << 9) + lane * 16;
    const v8h b0lo = *(const v8h*)bp0;
    const v8h b0hi = *(const v8h*)(bp0 + 8);
    const v8h b1lo = *(const v8h*)(bp0 + 512);
    const v8h b1hi = *(const v8h*)(bp0 + 520);

    const float4 a0 = *(const float4*)(arow + k0 + kHalfA);
    const float4 a1 = *(const float4*)(arow + k0 + kHalfA + 4);
    const float4 a2 = *(const float4*)(arow + k0 + kHalfA + 16);
    const float4 a3 = *(const float4*)(arow + k0 + kHalfA + 20);

    v16h a;
    a[0]  = (_Float16)a0.x; a[1]  = (_Float16)a0.y;
    a[2]  = (_Float16)a0.z; a[3]  = (_Float16)a0.w;
    a[4]  = (_Float16)a1.x; a[5]  = (_Float16)a1.y;
    a[6]  = (_Float16)a1.z; a[7]  = (_Float16)a1.w;
    a[8]  = (_Float16)a2.x; a[9]  = (_Float16)a2.y;
    a[10] = (_Float16)a2.z; a[11] = (_Float16)a2.w;
    a[12] = (_Float16)a3.x; a[13] = (_Float16)a3.y;
    a[14] = (_Float16)a3.z; a[15] = (_Float16)a3.w;

    v16h b0 = __builtin_shufflevector(b0lo, b0hi, 0,1,2,3,4,5,6,7,8,9,10,11,12,13,14,15);
    v16h b1 = __builtin_shufflevector(b1lo, b1hi, 0,1,2,3,4,5,6,7,8,9,10,11,12,13,14,15);

    acc0 = __builtin_amdgcn_wmma_f32_16x16x32_f16(
        false, a, false, b0, (short)0, acc0, false, false);
    acc1 = __builtin_amdgcn_wmma_f32_16x16x32_f16(
        false, a, false, b1, (short)0, acc1, false, false);
  }

  const int n0 = colBase + wn + (lane & 15);
  const int mb = (lane >> 4) * 8;
  const float bv0 = bias ? bias[n0] : 0.0f;
  const float bv1 = bias ? bias[n0 + 16] : 0.0f;
#pragma unroll
  for (int r = 0; r < 8; ++r) {
    int m = rowBase + wm + mb + r;
    float v0 = acc0[r] + bv0;
    float v1 = acc1[r] + bv1;
    if (act == 1)      { v0 = fmaxf(v0, 0.0f); v1 = fmaxf(v1, 0.0f); }
    else if (act == 2) { v0 = 1.0f / (1.0f + __expf(-v0));
                         v1 = 1.0f / (1.0f + __expf(-v1)); }
    C[(size_t)m * N + n0]      = v0;
    C[(size_t)m * N + n0 + 16] = v1;
  }
}

// --------------------------- small utility kernels -------------------------
__global__ void segnn_fill(float* __restrict__ p, float v, size_t n) {
  size_t i = (size_t)blockIdx.x * blockDim.x + threadIdx.x;
  if (i < n) p[i] = v;
}

__global__ void segnn_pos_concat(const float* __restrict__ x,
                                 const int* __restrict__ pos,
                                 const float* __restrict__ pos_emb,
                                 float* __restrict__ xin)
{
  __shared__ int   tg[8];
  __shared__ float cnt;
  int n = blockIdx.x, tid = threadIdx.x;
  if (tid < 8) tg[tid] = pos[n * 8 + tid];
  __syncthreads();
  if (tid == 0) {
    float c = 0.0f;
    for (int l = 0; l < 8; ++l) c += (tg[l] != 0) ? 1.0f : 0.0f;
    cnt = fmaxf(c, 1.0f);
  }
  __syncthreads();
  for (int j = tid; j < KPAD; j += 256) {
    float v;
    if (j < 768) v = x[(size_t)n * 768 + j];
    else if (j < 831) {
      int f = j - 768;
      float s = 0.0f;
      for (int l = 0; l < 8; ++l)
        if (tg[l] != 0) s += pos_emb[tg[l] * 63 + f];
      v = s / cnt;
    } else v = 0.0f;
    xin[(size_t)n * KPAD + j] = v;
  }
}

__global__ void segnn_gather_rows(const int* __restrict__ idx,
                                  const float* __restrict__ tbl,
                                  float* __restrict__ out)
{
  int e = blockIdx.x, f = threadIdx.x;
  out[(size_t)e * 128 + f] = tbl[(size_t)idx[e] * 128 + f];
}

__global__ void segnn_concat_g(const float* __restrict__ xn,
                               const int* __restrict__ src,
                               const int* __restrict__ dst,
                               const float* __restrict__ ef,
                               float* __restrict__ g)
{
  int e = blockIdx.x, f = threadIdx.x;
  int s = src[e], d = dst[e];
  size_t base = (size_t)e * 384;
  g[base + f]        = xn[(size_t)s * 128 + f];
  g[base + 128 + f]  = xn[(size_t)d * 128 + f];
  g[base + 256 + f]  = ef[(size_t)e * 128 + f];
}

__global__ void segnn_mul(float* __restrict__ ge,
                          const float* __restrict__ gates,
                          const float* __restrict__ ef, size_t n)
{
  size_t i = (size_t)blockIdx.x * blockDim.x + threadIdx.x;
  if (i < n) ge[i] = gates[i] * ef[i];
}

__global__ void segnn_scatter_agg(const float* __restrict__ ge,
                                  const int* __restrict__ src,
                                  const int* __restrict__ dst,
                                  float* __restrict__ agg)
{
  int e = blockIdx.x, f = threadIdx.x;
  float v = 0.5f * ge[(size_t)e * 128 + f];
  atomicAdd(&agg[(size_t)src[e] * 128 + f], v);
  atomicAdd(&agg[(size_t)dst[e] * 128 + f], v);
}

__global__ void segnn_add3(float* __restrict__ a, const float* __restrict__ b,
                           const float* __restrict__ c, size_t n)
{
  size_t i = (size_t)blockIdx.x * blockDim.x + threadIdx.x;
  if (i < n) a[i] = a[i] + b[i] + c[i];
}

__global__ void segnn_add2(float* __restrict__ o, const float* __restrict__ a,
                           const float* __restrict__ b, size_t n)
{
  size_t i = (size_t)blockIdx.x * blockDim.x + threadIdx.x;
  if (i < n) o[i] = a[i] + b[i];
}

// --------------------- TransformerConv attention path ----------------------
__global__ __launch_bounds__(128) void segnn_tc_alpha(
    const float* __restrict__ q, const float* __restrict__ k,
    const float* __restrict__ e, const int* __restrict__ src,
    const int* __restrict__ dst, float* __restrict__ alpha,
    float* __restrict__ segmax)
{
  int eid = blockIdx.x;
  int head = threadIdx.x >> 5, lane = threadIdx.x & 31;
  int s = src[eid], d = dst[eid];
  float sum = 0.0f;
#pragma unroll
  for (int t = 0; t < 4; ++t) {
    int j = lane + 32 * t;
    float qv = q[(size_t)d * 512 + head * 128 + j];
    float kv = k[(size_t)s * 512 + head * 128 + j] +
               e[(size_t)eid * 512 + head * 128 + j];
    sum += qv * kv;
  }
  sum = wave_reduce_sum(sum);
  if (lane == 0) {
    float a = sum * 0.08838834764831845f;
    alpha[(size_t)eid * 4 + head] = a;
    atomicMaxFloat(&segmax[(size_t)d * 4 + head], a);
  }
}

__global__ void segnn_tc_exp(float* __restrict__ alpha,
                             const int* __restrict__ dst,
                             const float* __restrict__ segmax,
                             float* __restrict__ segsum, size_t n)
{
  size_t i = (size_t)blockIdx.x * blockDim.x + threadIdx.x;
  if (i >= n) return;
  size_t eid = i >> 2; int head = (int)(i & 3);
  int d = dst[eid];
  float ex = __expf(alpha[i] - segmax[(size_t)d * 4 + head]);
  alpha[i] = ex;
  atomicAdd(&segsum[(size_t)d * 4 + head], ex);
}

__global__ __launch_bounds__(128) void segnn_tc_scatter(
    const float* __restrict__ alpha, const float* __restrict__ segsum,
    const float* __restrict__ v, const float* __restrict__ e,
    const int* __restrict__ src, const int* __restrict__ dst,
    float* __restrict__ outacc)
{
  int eid = blockIdx.x, f = threadIdx.x;
  int s = src[eid], d = dst[eid];
#pragma unroll
  for (int head = 0; head < 4; ++head) {
    float w = alpha[(size_t)eid * 4 + head] /
              (segsum[(size_t)d * 4 + head] + 1e-16f);
    float val = (v[(size_t)s * 512 + head * 128 + f] +
                 e[(size_t)eid * 512 + head * 128 + f]) * w;
    atomicAdd(&outacc[((size_t)d * 4 + head) * 128 + f], val);
  }
}

__global__ void segnn_tc_combine(const float* __restrict__ outacc,
                                 const float* __restrict__ skip,
                                 float* __restrict__ xn, size_t n)
{
  size_t i = (size_t)blockIdx.x * blockDim.x + threadIdx.x;
  if (i >= n) return;
  size_t node = i >> 7; int f = (int)(i & 127);
  float s = 0.0f;
#pragma unroll
  for (int head = 0; head < 4; ++head)
    s += outacc[((size_t)node * 4 + head) * 128 + f];
  xn[i] = 0.25f * s + skip[i];
}

// --------------------------- GATv2 attention path --------------------------
__global__ __launch_bounds__(128) void segnn_gat_alpha(
    const float* __restrict__ xl, const float* __restrict__ xr,
    const float* __restrict__ att, const int* __restrict__ src,
    const int* __restrict__ dst, float* __restrict__ alpha,
    float* __restrict__ segmax, int E)
{
  int i = blockIdx.x;
  int s, d;
  if (i < E) { s = src[i]; d = dst[i]; } else { s = d = i - E; }
  int head = threadIdx.x >> 5, lane = threadIdx.x & 31;
  float sum = 0.0f;
#pragma unroll
  for (int t = 0; t < 4; ++t) {
    int j = lane + 32 * t;
    float z = xl[(size_t)s * 512 + head * 128 + j] +
              xr[(size_t)d * 512 + head * 128 + j];
    z = (z < 0.0f) ? 0.2f * z : z;
    sum += z * att[head * 128 + j];
  }
  sum = wave_reduce_sum(sum);
  if (lane == 0) {
    alpha[(size_t)i * 4 + head] = sum;
    atomicMaxFloat(&segmax[(size_t)d * 4 + head], sum);
  }
}

__global__ void segnn_gat_exp(float* __restrict__ alpha,
                              const int* __restrict__ dst,
                              const float* __restrict__ segmax,
                              float* __restrict__ segsum, int E, size_t n)
{
  size_t i = (size_t)blockIdx.x * blockDim.x + threadIdx.x;
  if (i >= n) return;
  size_t idx = i >> 2; int head = (int)(i & 3);
  int d = (idx < (size_t)E) ? dst[idx] : (int)(idx - E);
  float ex = __expf(alpha[i] - segmax[(size_t)d * 4 + head]);
  alpha[i] = ex;
  atomicAdd(&segsum[(size_t)d * 4 + head], ex);
}

__global__ __launch_bounds__(128) void segnn_gat_scatter(
    const float* __restrict__ alpha, const float* __restrict__ segsum,
    const float* __restrict__ xl, const int* __restrict__ src,
    const int* __restrict__ dst, float* __restrict__ outacc, int E)
{
  int i = blockIdx.x, f = threadIdx.x;
  int s, d;
  if (i < E) { s = src[i]; d = dst[i]; } else { s = d = i - E; }
#pragma unroll
  for (int head = 0; head < 4; ++head) {
    float w = alpha[(size_t)i * 4 + head] /
              (segsum[(size_t)d * 4 + head] + 1e-16f);
    atomicAdd(&outacc[((size_t)d * 4 + head) * 128 + f],
              xl[(size_t)s * 512 + head * 128 + f] * w);
  }
}

__global__ void segnn_gat_combine(const float* __restrict__ outacc,
                                  const float* __restrict__ bias,
                                  float* __restrict__ xn, size_t n)
{
  size_t i = (size_t)blockIdx.x * blockDim.x + threadIdx.x;
  if (i >= n) return;
  size_t node = i >> 7; int f = (int)(i & 127);
  float s = 0.0f;
#pragma unroll
  for (int head = 0; head < 4; ++head)
    s += outacc[((size_t)node * 4 + head) * 128 + f];
  xn[i] = 0.25f * s + bias[f];
}

// ------------------------------- GraphNorm ---------------------------------
__global__ __launch_bounds__(128) void segnn_graphnorm_relu(
    float* __restrict__ xn, const float* __restrict__ w,
    const float* __restrict__ b, const float* __restrict__ ms)
{
  int g = blockIdx.x, f = threadIdx.x;
  size_t base = (size_t)g * NG_CONST * 128;
  float s = 0.0f;
  for (int i = 0; i < NG_CONST; ++i) s += xn[base + (size_t)i * 128 + f];
  float sub = (s * (1.0f / NG_CONST)) * ms[f];
  float var = 0.0f;
  for (int i = 0; i < NG_CONST; ++i) {
    float o = xn[base + (size_t)i * 128 + f] - sub;
    var += o * o;
  }
  float istd = rsqrtf(var * (1.0f / NG_CONST) + 1e-5f);
  float wf = w[f], bf = b[f];
  for (int i = 0; i < NG_CONST; ++i) {
    float o = xn[base + (size_t)i * 128 + f] - sub;
    xn[base + (size_t)i * 128 + f] = fmaxf(wf * o * istd + bf, 0.0f);
  }
}

// ----------------------------- MHA pooling ---------------------------------
__global__ __launch_bounds__(128) void segnn_mha_pool(
    const float* __restrict__ hfin, const float* __restrict__ gq,
    const float* __restrict__ in_w, const float* __restrict__ in_b,
    const float* __restrict__ out_w, const float* __restrict__ out_b,
    float* __restrict__ pooled)
{
  __shared__ float qp[128];
  __shared__ float kv[128];
  __shared__ float sc[4 * NG_CONST];
  __shared__ float red[4];
  int g = blockIdx.x, o = threadIdx.x;

  float s = in_b[o];
  for (int i = 0; i < 128; ++i) s += gq[i] * in_w[(size_t)o * 128 + i];
  qp[o] = s;
  __syncthreads();

  const float rs = 0.17677669529663687f;
  for (int t = 0; t < NG_CONST; ++t) {
    const float* hrow = hfin + ((size_t)g * NG_CONST + t) * 128;
    float kk = in_b[128 + o];
    for (int i = 0; i < 128; ++i) kk += hrow[i] * in_w[(size_t)(128 + o) * 128 + i];
    kv[o] = kk;
    __syncthreads();
    if (o < 4) {
      float d = 0.0f;
      for (int hd = 0; hd < 32; ++hd) d += qp[o * 32 + hd] * kv[o * 32 + hd];
      sc[o * NG_CONST + t] = d * rs;
    }
    __syncthreads();
  }

  if (o < 4) {
    float mx = -1e30f;
    for (int t = 0; t < NG_CONST; ++t) mx = fmaxf(mx, sc[o * NG_CONST + t]);
    float sm = 0.0f;
    for (int t = 0; t < NG_CONST; ++t) {
      float e = __expf(sc[o * NG_CONST + t] - mx);
      sc[o * NG_CONST + t] = e;
      sm += e;
    }
    red[o] = sm;
  }
  __syncthreads();

  int head = o >> 5;
  float inv = 1.0f / red[head];
  float acc = 0.0f;
  for (int t = 0; t < NG_CONST; ++t) {
    const float* hrow = hfin + ((size_t)g * NG_CONST + t) * 128;
    float vv = in_b[256 + o];
    for (int i = 0; i < 128; ++i) vv += hrow[i] * in_w[(size_t)(256 + o) * 128 + i];
    acc += sc[head * NG_CONST + t] * inv * vv;
  }
  __syncthreads();
  kv[o] = acc;
  __syncthreads();
  float po = out_b[o];
  for (int i = 0; i < 128; ++i) po += kv[i] * out_w[(size_t)o * 128 + i];
  pooled[(size_t)g * 128 + o] = po;
}

// --------------------------- final head + losses ---------------------------
__global__ __launch_bounds__(64) void segnn_head(
    const float* __restrict__ pooled, const int* __restrict__ y,
    const float* __restrict__ fcW, const float* __restrict__ fcb,
    const float* __restrict__ fwW, const float* __restrict__ fwb,
    float* __restrict__ out)
{
  __shared__ float Om[5][5];
  __shared__ float colsum[5];
  __shared__ float ohcnt[5];
  __shared__ float coral_s[64];
  int g = threadIdx.x;
  if (g < 5) { colsum[g] = 0.0f; ohcnt[g] = 0.0f; }
  if (g < 25) Om[g / 5][g % 5] = 0.0f;
  __syncthreads();

  const float* p = pooled + (size_t)g * 128;
  float mu = 0.0f;
  for (int i = 0; i < 128; ++i) mu += p[i];
  mu *= (1.0f / 128.0f);
  float var = 0.0f;
  for (int i = 0; i < 128; ++i) { float d = p[i] - mu; var += d * d; }
  var *= (1.0f / 128.0f);
  float inv = rsqrtf(var + 1e-5f);

  float lg[4];
  for (int c = 0; c < 4; ++c) {
    float s = fcb[c];
    for (int i = 0; i < 128; ++i) s += ((p[i] - mu) * inv) * fcW[i * 4 + c];
    lg[c] = s;
    out[1 + g * 4 + c] = s;
  }
  int yy = y[g];
  float cor = 0.0f;
  for (int c = 0; c < 4; ++c) {
    float lv = (c < yy) ? 1.0f : 0.0f;
    float ls = logsigf(lg[c]);
    cor -= ls * lv + (ls - lg[c]) * (1.0f - lv);
  }
  coral_s[g] = cor;

  float lw[5];
  float mx = -1e30f;
  for (int c = 0; c < 5; ++c) {
    float s = fwb[c];
    for (int i = 0; i < 128; ++i) s += ((p[i] - mu) * inv) * fwW[i * 5 + c];
    lw[c] = s;
    mx = fmaxf(mx, s);
  }
  float sm = 0.0f;
  for (int c = 0; c < 5; ++c) { lw[c] = __expf(lw[c] - mx); sm += lw[c]; }
  float psum = 0.0f;
  for (int c = 0; c < 5; ++c) { lw[c] /= sm; psum += lw[c]; }
  for (int c = 0; c < 5; ++c) {
    float yp = lw[c] / (psum + 1e-10f);
    atomicAdd(&Om[c][yy], yp);
    atomicAdd(&colsum[c], yp);
  }
  atomicAdd(&ohcnt[yy], 1.0f);
  __syncthreads();

  if (g == 0) {
    float cm = 0.0f;
    for (int i = 0; i < 64; ++i) cm += coral_s[i];
    cm *= (1.0f / 64.0f);
    float wnum = 0.0f, wden = 0.0f;
    for (int i = 0; i < 5; ++i)
      for (int j = 0; j < 5; ++j) {
        float w = (float)((i - j) * (i - j)) * (1.0f / 16.0f);
        wnum += w * Om[i][j];
        wden += w * (colsum[i] * ohcnt[j] * (1.0f / 64.0f));
      }
    float kappa = 1.0f - wnum / (wden + 1e-10f);
    out[0] = 0.2f * cm + 0.8f * logf(1.0f - kappa + 1e-10f);
  }
}

// ---------------------------------------------------------------------------
extern "C" void kernel_launch(void* const* d_in, const int* in_sizes, int n_in,
                              void* d_out, int out_size, void* d_ws, size_t ws_size,
                              hipStream_t stream)
{
  (void)in_sizes; (void)n_in; (void)out_size; (void)ws_size;
  constexpr int NN = NN_CONST, EE = E_CONST, GG = G_CONST;
  constexpr int EN = EE + NN;

  const float* x     = (const float*)d_in[0];
  const int*   y     = (const int*)d_in[1];
  const int*   eidx  = (const int*)d_in[2];
  const int*   eattr = (const int*)d_in[3];
  const int*   pos   = (const int*)d_in[5];
  const int*   src   = eidx;
  const int*   dst   = eidx + EE;

  auto F = [&](int i) { return (const float*)d_in[i]; };
  const float* pos_emb  = F(6);
  const float* edge_emb = F(7);
  const float *epW1 = F(8),  *epb1 = F(9),  *epW2 = F(10), *epb2 = F(11);
  const float *npW1 = F(12), *npb1 = F(13), *npW2 = F(14), *npb2 = F(15);
  const float *egW1 = F(16), *egb1 = F(17), *egW2 = F(18), *egb2 = F(19);
  const float *efW  = F(20), *efb  = F(21);

  struct EvenP { const float *Wq,*bq,*Wk,*bk,*Wv,*bv,*We,*Wskip,*bskip,*nw,*nb,*nms; };
  struct OddP  { const float *Wl,*bl,*Wr,*br,*att,*bias,*nw,*nb,*nms; };
  EvenP ev[2]; OddP od[2];
  int pi = 22;
  for (int li = 0; li < 4; ++li) {
    if ((li & 1) == 0) {
      EvenP& p = ev[li / 2];
      p.Wq = F(pi++); p.bq = F(pi++); p.Wk = F(pi++); p.bk = F(pi++);
      p.Wv = F(pi++); p.bv = F(pi++); p.We = F(pi++);
      p.Wskip = F(pi++); p.bskip = F(pi++);
      p.nw = F(pi++); p.nb = F(pi++); p.nms = F(pi++);
    } else {
      OddP& p = od[li / 2];
      p.Wl = F(pi++); p.bl = F(pi++); p.Wr = F(pi++); p.br = F(pi++);
      p.att = F(pi++); p.bias = F(pi++);
      p.nw = F(pi++); p.nb = F(pi++); p.nms = F(pi++);
    }
  }
  const float* gq  = F(pi++);
  const float *apw = F(pi++), *apb = F(pi++), *apow = F(pi++), *apob = F(pi++);
  const float *fcW = F(pi++), *fcb = F(pi++);
  const float *fwW = F(pi++), *fwb = F(pi++);

  // ---- workspace bump allocator ----
  char* wp = (char*)d_ws;
  auto alloc = [&](size_t elems) {
    float* p = (float*)wp;
    wp += ((elems * sizeof(float) + 255) / 256) * 256;
    return p;
  };
  auto allocH = [&](size_t elems) {
    _Float16* p = (_Float16*)wp;
    wp += ((elems * sizeof(_Float16) + 255) / 256) * 256;
    return p;
  };

  auto gridFor = [](size_t n) { return (unsigned)((n + 255) / 256); };

  auto swz = [&](const float* W, int Ksrc, int Kdst, int N) {
    _Float16* o = allocH((size_t)Kdst * N);
    segnn_swizzle_w<<<gridFor((size_t)Kdst * N), 256, 0, stream>>>(
        W, o, Ksrc, N, (size_t)Kdst * N);
    return o;
  };
  _Float16* s_npW1 = swz(npW1, 831, KPAD, 128);
  _Float16* s_npW2 = swz(npW2, 128, 128, 128);
  _Float16* s_epW1 = swz(epW1, 128, 128, 128);
  _Float16* s_epW2 = swz(epW2, 128, 128, 128);
  _Float16* s_egW1 = swz(egW1, 384, 384, 128);
  _Float16* s_egW2 = swz(egW2, 128, 128, 128);
  _Float16* s_efW  = swz(efW,  128, 128, 128);
  _Float16 *s_Wq[2], *s_Wk[2], *s_Wv[2], *s_We[2], *s_Wskip[2];
  _Float16 *s_Wl[2], *s_Wr[2];
  for (int i = 0; i < 2; ++i) {
    s_Wq[i]    = swz(ev[i].Wq,    128, 128, 512);
    s_Wk[i]    = swz(ev[i].Wk,    128, 128, 512);
    s_Wv[i]    = swz(ev[i].Wv,    128, 128, 512);
    s_We[i]    = swz(ev[i].We,    128, 128, 512);
    s_Wskip[i] = swz(ev[i].Wskip, 128, 128, 128);
    s_Wl[i]    = swz(od[i].Wl,    128, 128, 512);
    s_Wr[i]    = swz(od[i].Wr,    128, 128, 512);
  }

  float* xin    = alloc((size_t)NN * KPAD);
  float* hbuf   = alloc((size_t)NN * 128);
  float* hres   = alloc((size_t)NN * 128);
  float* xprev  = alloc((size_t)NN * 128);
  float* xn     = alloc((size_t)NN * 128);
  float* tmp    = alloc((size_t)NN * 128);
  float* ef0    = alloc((size_t)EE * 128);
  float* efh    = alloc((size_t)EE * 128);
  float* ef     = alloc((size_t)EE * 128);
  float* qb     = alloc((size_t)NN * 512);
  float* kb     = alloc((size_t)NN * 512);
  float* vb     = alloc((size_t)NN * 512);
  float* ebuf   = alloc((size_t)EE * 512);
  float* outacc = alloc((size_t)NN * 512);
  float* alpha  = alloc((size_t)EN * 4);
  float* segmax = alloc((size_t)NN * 4);
  float* segsum = alloc((size_t)NN * 4);
  float* pooled = alloc((size_t)GG * 128);

  auto fillk = [&](float* p, float v, size_t n) {
    segnn_fill<<<gridFor(n), 256, 0, stream>>>(p, v, n);
  };
  auto gemm = [&](const float* A, const _Float16* Wsw, const float* bias, float* C,
                  int M, int K, int N, int act) {
    dim3 grid((unsigned)((M / 32) * (N / 128)));
    if (K == 128)
      segnn_gemm_wmma_k128<<<grid, 256, 0, stream>>>(A, Wsw, bias, C, M, N, act);
    else
      segnn_gemm_wmma<<<grid, 256, 0, stream>>>(A, Wsw, bias, C, M, K, N, act);
  };

  // ---- node features ----
  segnn_pos_concat<<<NN, 256, 0, stream>>>(x, pos, pos_emb, xin);
  gemm(xin, s_npW1, npb1, hbuf, NN, KPAD, 128, 1);
  gemm(hbuf, s_npW2, npb2, hres, NN, 128, 128, 1);
  hipMemcpyAsync(xprev, hres, sizeof(float) * (size_t)NN * 128,
                 hipMemcpyDeviceToDevice, stream);

  // ---- edge features ----
  segnn_gather_rows<<<EE, 128, 0, stream>>>(eattr, edge_emb, ef0);
  gemm(ef0, s_epW1, epb1, efh, EE, 128, 128, 1);
  gemm(efh, s_epW2, epb2, ef, EE, 128, 128, 0);

  auto agg_edge = [&]() {
    float* gcat  = ebuf;
    float* gateh = ef0;
    float* gates = efh;
    float* ge    = ef0;
    float* agg   = outacc;
    segnn_concat_g<<<EE, 128, 0, stream>>>(xn, src, dst, ef, gcat);
    gemm(gcat, s_egW1, egb1, gateh, EE, 384, 128, 1);
    gemm(gateh, s_egW2, egb2, gates, EE, 128, 128, 2);
    segnn_mul<<<gridFor((size_t)EE * 128), 256, 0, stream>>>(ge, gates, ef,
                                                             (size_t)EE * 128);
    fillk(agg, 0.0f, (size_t)NN * 128);
    segnn_scatter_agg<<<EE, 128, 0, stream>>>(ge, src, dst, agg);
    gemm(agg, s_efW, efb, tmp, NN, 128, 128, 0);
    segnn_add3<<<gridFor((size_t)NN * 128), 256, 0, stream>>>(
        xprev, xn, tmp, (size_t)NN * 128);
  };

  for (int li = 0; li < 4; ++li) {
    if ((li & 1) == 0) {
      const EvenP& p = ev[li / 2];
      gemm(xprev, s_Wq[li / 2], p.bq, qb, NN, 128, 512, 0);
      gemm(xprev, s_Wk[li / 2], p.bk, kb, NN, 128, 512, 0);
      gemm(xprev, s_Wv[li / 2], p.bv, vb, NN, 128, 512, 0);
      gemm(ef, s_We[li / 2], nullptr, ebuf, EE, 128, 512, 0);
      gemm(xprev, s_Wskip[li / 2], p.bskip, tmp, NN, 128, 128, 0);
      fillk(segmax, -1e30f, (size_t)NN * 4);
      fillk(segsum, 0.0f, (size_t)NN * 4);
      fillk(outacc, 0.0f, (size_t)NN * 512);
      segnn_tc_alpha<<<EE, 128, 0, stream>>>(qb, kb, ebuf, src, dst, alpha, segmax);
      segnn_tc_exp<<<gridFor((size_t)EE * 4), 256, 0, stream>>>(
          alpha, dst, segmax, segsum, (size_t)EE * 4);
      segnn_tc_scatter<<<EE, 128, 0, stream>>>(alpha, segsum, vb, ebuf, src, dst,
                                               outacc);
      segnn_tc_combine<<<gridFor((size_t)NN * 128), 256, 0, stream>>>(
          outacc, tmp, xn, (size_t)NN * 128);
      segnn_graphnorm_relu<<<GG, 128, 0, stream>>>(xn, p.nw, p.nb, p.nms);
    } else {
      const OddP& p = od[li / 2];
      gemm(xprev, s_Wl[li / 2], p.bl, qb, NN, 128, 512, 0);
      gemm(xprev, s_Wr[li / 2], p.br, kb, NN, 128, 512, 0);
      fillk(segmax, -1e30f, (size_t)NN * 4);
      fillk(segsum, 0.0f, (size_t)NN * 4);
      fillk(outacc, 0.0f, (size_t)NN * 512);
      segnn_gat_alpha<<<EN, 128, 0, stream>>>(qb, kb, p.att, src, dst, alpha,
                                              segmax, EE);
      segnn_gat_exp<<<gridFor((size_t)EN * 4), 256, 0, stream>>>(
          alpha, dst, segmax, segsum, EE, (size_t)EN * 4);
      segnn_gat_scatter<<<EN, 128, 0, stream>>>(alpha, segsum, qb, src, dst,
                                                outacc, EE);
      segnn_gat_combine<<<gridFor((size_t)NN * 128), 256, 0, stream>>>(
          outacc, p.bias, xn, (size_t)NN * 128);
      segnn_graphnorm_relu<<<GG, 128, 0, stream>>>(xn, p.nw, p.nb, p.nms);
    }
    agg_edge();
  }

  // ---- readout ----
  segnn_add2<<<gridFor((size_t)NN * 128), 256, 0, stream>>>(
      hbuf, xprev, hres, (size_t)NN * 128);
  segnn_mha_pool<<<GG, 128, 0, stream>>>(hbuf, gq, apw, apb, apow, apob, pooled);
  segnn_head<<<1, 64, 0, stream>>>(pooled, y, fcW, fcb, fwW, fwb, (float*)d_out);
}
